// FHNIntegrator_63642825392409
// MI455X (gfx1250) — matrix operand order repaired and checked by reading, not compiled, not run on hardware
//
#include <hip/hip_runtime.h>
#include <stdint.h>

// ---------------------------------------------------------------------------
// FitzHugh-Nagumo network RK4 integrator for MI455X (gfx1250).
//
// Per timestep: 4 RK4 stages. Each stage = one fused kernel:
//   icouple = tanh(q_eval) @ J^T        (WMMA f16 -> f32 accumulate)
//   epilogue: dq/dp, RK4 accumulation, next q_eval/p_eval, next tanh->f16,
//             stage 4: clamp + state update + trace write.
//
// Global->LDS tile movement uses the CDNA5 Tensor Data Mover (TDM) with the
// descriptor pad feature producing a padded LDS pitch, double-buffered so the
// DMA of chunk k+1 overlaps the WMMA work on chunk k. Loop-invariant
// descriptor state is hoisted out of the K-loop. Fallback (if the TDM builtin
// is unavailable) is a vectorized b128 cooperative copy.
// ---------------------------------------------------------------------------

typedef _Float16 v16h __attribute__((ext_vector_type(16)));
typedef float    v8f  __attribute__((ext_vector_type(8)));
typedef uint32_t v4u  __attribute__((ext_vector_type(4)));
typedef int      v4i  __attribute__((ext_vector_type(4)));
typedef int      v8i  __attribute__((ext_vector_type(8)));

#define F_DT     0.01f
#define F_EPS    0.08f
#define F_ALPHA  0.7f
#define F_BETA   0.8f
#define F_CLAMP  20.0f

#define NB 256    // batch
#define NT 128    // timesteps
#define NN 2048   // neurons

#define BM 64     // block tile M (batch)
#define BN 128    // block tile N (neuron)
#define BK 64     // k chunk (halves)
#define LDK 36    // LDS row pitch in dwords (32 data + 4 pad); 144B = 9*16B

#define LDSA_DW 0                   // A tile offset (dwords) within a buffer
#define LDSB_DW (BM * LDK)          // 2304
#define BUF_DW  (BM * LDK + BN * LDK)   // 6912 dwords = 27648 bytes / buffer

#if __has_builtin(__builtin_amdgcn_tensor_load_to_lds) && \
    __has_builtin(__builtin_amdgcn_s_wait_tensorcnt)
#define USE_TDM 1
#else
#define USE_TDM 0
#endif

union AFrag { uint32_t u[8]; v16h v; };
union CFrag { float    f[8]; v8f  v; };

#if USE_TDM
// Build the loop-invariant D# group 1: 2-byte elements, LDS padding of 4
// dwords after every 32 dwords (pitch 36 dw), 2D tile BK x tile_rows,
// row stride NN elements, huge tensor dims (tiles are always in-bounds).
__device__ __forceinline__ v8i tdm_make_g1(uint32_t tile_rows) {
  v8i g1;
  g1[0] = (int)((1u << 16)                      // data_size = 2 bytes
              | (1u << 20)                      // pad_enable
              | (4u << 22)                      // pad_interval code 4 = 32 dw
              | (3u << 25));                    // pad_amount  code 3 = 4 dw
  g1[1] = 0;                                    // no atomic barrier; dim0 lo16=0
  g1[2] = 0x10;                                 // tensor_dim0 = 1M (hi16); dim1 lo16=0
  g1[3] = (int)(0x10u | ((uint32_t)BK << 16));  // tensor_dim1 = 1M; tile_dim0 = 64
  g1[4] = (int)(tile_rows & 0xFFFFu);           // tile_dim1; tile_dim2 = 0
  g1[5] = (int)NN;                              // tensor_dim0_stride = 2048 elems
  g1[6] = 0;
  g1[7] = 0;
  return g1;
}

__device__ __forceinline__ void tdm_issue(uint32_t lds_byte_off, uint64_t ga,
                                          v8i g1) {
  v4u g0;
  g0.x = 1u;                                    // count=1 (valid user D#)
  g0.y = lds_byte_off;                          // LDS byte address
  g0.z = (uint32_t)ga;                          // global addr [31:0]
  g0.w = (uint32_t)((ga >> 32) & 0x01FFFFFFu)   // global addr [56:32]
       | (2u << 30);                            // type = 2 ("image")
  v4i z4 = {0, 0, 0, 0};
#if __clang_major__ >= 23
  v8i z8 = {0, 0, 0, 0, 0, 0, 0, 0};
  __builtin_amdgcn_tensor_load_to_lds(g0, g1, z4, z4, z8, 0);
#else
  __builtin_amdgcn_tensor_load_to_lds(g0, g1, z4, z4, 0);
#endif
}
#endif

// --- setup kernels ---------------------------------------------------------

__global__ __launch_bounds__(256)
void k_cvt_coupling(const float* __restrict__ J, _Float16* __restrict__ Jh) {
  int i = (blockIdx.x * 256 + threadIdx.x) * 4;
  #pragma unroll
  for (int k = 0; k < 4; ++k) Jh[i + k] = (_Float16)J[i + k];
}

__global__ __launch_bounds__(256)
void k_init(const float* __restrict__ q0, const float* __restrict__ p0,
            float* __restrict__ q,  float* __restrict__ p,
            float* __restrict__ qe, float* __restrict__ pe,
            _Float16* __restrict__ X) {
  int i = blockIdx.x * 256 + threadIdx.x;
  float qv = q0[i], pv = p0[i];
  q[i] = qv; p[i] = pv; qe[i] = qv; pe[i] = pv;
  X[i] = (_Float16)tanhf(qv);
}

// --- fused GEMM + RK4 stage ------------------------------------------------

template <int STAGE>
__global__ __launch_bounds__(256)
void k_fhn_stage(const _Float16* __restrict__ X,   // [NB][NN] f16 tanh(q_eval)
                 const _Float16* __restrict__ Jh,  // [NN][NN] f16 coupling
                 const float* __restrict__ drive,  // [NB][NT][NN]
                 const float* __restrict__ bias,   // [NN]
                 float* __restrict__ q,  float* __restrict__ p,
                 float* __restrict__ qe, float* __restrict__ pe,
                 float* __restrict__ qa, float* __restrict__ pa,
                 _Float16* __restrict__ Xn,        // f16 out for next stage
                 float* __restrict__ trace,        // [NB][NT][NN]
                 int t)
{
  __shared__ uint32_t lds[2 * BUF_DW];             // double-buffered tiles

  const int tid   = threadIdx.x;
  const int lane  = tid & 31;
  const int wv    = tid >> 5;      // 0..7
  const int g     = lane >> 4;     // lane half-group
  const int ml    = lane & 15;
  const int waveM = wv >> 2;       // 0..1  (32 rows each)
  const int waveN = wv & 3;        // 0..3  (32 cols each)

  const int blockN = blockIdx.x * BN;
  const int blockM = blockIdx.y * BM;

  CFrag acc[2][2];
  #pragma unroll
  for (int mt = 0; mt < 2; ++mt)
    #pragma unroll
    for (int nt = 0; nt < 2; ++nt)
      acc[mt][nt].v = (v8f){0.f,0.f,0.f,0.f,0.f,0.f,0.f,0.f};

#if USE_TDM
  // loop-invariant TDM state
  const v8i g1A = tdm_make_g1(BM);
  const v8i g1B = tdm_make_g1(BN);
  const uint64_t gaA = (uint64_t)(uintptr_t)(X  + (size_t)blockM * NN);
  const uint64_t gaB = (uint64_t)(uintptr_t)(Jh + (size_t)blockN * NN);
  // issue chunk 0 into buffer 0
  if (wv == 0) {
    tdm_issue(0u,            gaA, g1A);
    tdm_issue(LDSB_DW * 4u,  gaB, g1B);
  }
#else
  auto issue_chunk = [&](int buf, int k0) {
    const uint4* XgV = (const uint4*)X;       // row = NN/8 uint4
    const uint4* JgV = (const uint4*)Jh;
    uint4* sA = (uint4*)&lds[buf * BUF_DW + LDSA_DW];
    uint4* sB = (uint4*)&lds[buf * BUF_DW + LDSB_DW];
    const int kq = k0 >> 3;                   // chunk offset in uint4
    uint4 ta[2], tb[4];
    #pragma unroll
    for (int i = 0; i < 2; ++i) {             // A: 512 uint4, 2/thread
      int idx = tid + i * 256;
      ta[i] = XgV[(size_t)(blockM + (idx >> 3)) * (NN/8) + kq + (idx & 7)];
    }
    #pragma unroll
    for (int i = 0; i < 4; ++i) {             // B: 1024 uint4, 4/thread
      int idx = tid + i * 256;
      tb[i] = JgV[(size_t)(blockN + (idx >> 3)) * (NN/8) + kq + (idx & 7)];
    }
    #pragma unroll
    for (int i = 0; i < 2; ++i) {
      int idx = tid + i * 256;
      sA[(idx >> 3) * 9 + (idx & 7)] = ta[i]; // pitch 36 dw = 9 uint4
    }
    #pragma unroll
    for (int i = 0; i < 4; ++i) {
      int idx = tid + i * 256;
      sB[(idx >> 3) * 9 + (idx & 7)] = tb[i];
    }
  };
  issue_chunk(0, 0);
#endif

  int cur = 0;
  for (int k0 = 0; k0 < NN; k0 += BK) {
#if USE_TDM
    // scalar wait; no-op on waves that issued nothing
    __builtin_amdgcn_s_wait_tensorcnt(0);
#endif
    __syncthreads();                          // buffer `cur` is ready
    if (k0 + BK < NN) {
#if USE_TDM
      if (wv == 0) {
        const uint32_t base = (uint32_t)((cur ^ 1) * (BUF_DW * 4));
        const uint64_t koff = (uint64_t)(uint32_t)((k0 + BK) * 2);
        tdm_issue(base,                gaA + koff, g1A);
        tdm_issue(base + LDSB_DW * 4u, gaB + koff, g1B);
      }
#else
      issue_chunk(cur ^ 1, k0 + BK);
#endif
    }

    const uint32_t* AsU = &lds[cur * BUF_DW + LDSA_DW];
    const uint32_t* BsU = &lds[cur * BUF_DW + LDSB_DW];

    #pragma unroll
    for (int ks = 0; ks < BK; ks += 32) {
      AFrag a[2], b[2];
      // A fragment 16x32 f16: lanes 0-15 M=0-15; v0-3: K=8g+2v, v4-7: K=16+8g+2(v-4)
      #pragma unroll
      for (int mt = 0; mt < 2; ++mt) {
        int rowA = waveM * 32 + mt * 16 + ml;
        #pragma unroll
        for (int v = 0; v < 8; ++v) {
          int kIdx = (v < 4) ? (8 * g + 2 * v) : (16 + 8 * g + 2 * (v - 4));
          a[mt].u[v] = AsU[rowA * LDK + ((ks + kIdx) >> 1)];
        }
      }
      // B fragment 32x16 f16 (KxN): v0-3: K=8g+2v, v4-7: K=16+8g+2(v-4), B[k][n]=J[blockN+n][k]
      #pragma unroll
      for (int nt = 0; nt < 2; ++nt) {
        int rowB = waveN * 32 + nt * 16 + ml;
        #pragma unroll
        for (int v = 0; v < 8; ++v) {
          int kIdx = (v < 4) ? (8 * g + 2 * v) : (16 + 8 * g + 2 * (v - 4));
          b[nt].u[v] = BsU[rowB * LDK + ((ks + kIdx) >> 1)];
        }
      }
      #pragma unroll
      for (int mt = 0; mt < 2; ++mt)
        #pragma unroll
        for (int nt = 0; nt < 2; ++nt)
          acc[mt][nt].v = __builtin_amdgcn_wmma_f32_16x16x32_f16(
              false, a[mt].v, false, b[nt].v, (short)0, acc[mt][nt].v,
              false, false);
    }
    cur ^= 1;
  }

  // ---- fused RK4-stage epilogue ----
  // C layout: lanes 0-15 N=lane, M=r; lanes 16-31 N=lane-16, M=8+r
  #pragma unroll
  for (int mt = 0; mt < 2; ++mt) {
    #pragma unroll
    for (int nt = 0; nt < 2; ++nt) {
      const int n = blockN + waveN * 32 + nt * 16 + ml;
      const float bn = bias[n];
      #pragma unroll
      for (int r = 0; r < 8; ++r) {
        const int b = blockM + waveM * 32 + mt * 16 + g * 8 + r;
        const size_t e  = (size_t)b * NN + n;
        const size_t et = ((size_t)b * NT + t) * NN + n;

        const float ic  = acc[mt][nt].f[r] + bn;
        const float qev = qe[e];
        const float pev = pe[e];
        const float dq  = qev - (qev * qev * qev) * (1.0f / 3.0f) - pev
                        + drive[et] + ic;
        const float dp  = F_EPS * (qev + F_ALPHA - F_BETA * pev);

        if (STAGE == 1) {
          qa[e] = dq;                 pa[e] = dp;
          float qn2 = q[e] + 0.5f * F_DT * dq;
          float pn2 = p[e] + 0.5f * F_DT * dp;
          qe[e] = qn2; pe[e] = pn2;
          Xn[e] = (_Float16)tanhf(qn2);
        } else if (STAGE == 2) {
          qa[e] += 2.0f * dq;         pa[e] += 2.0f * dp;
          float qn2 = q[e] + 0.5f * F_DT * dq;
          float pn2 = p[e] + 0.5f * F_DT * dp;
          qe[e] = qn2; pe[e] = pn2;
          Xn[e] = (_Float16)tanhf(qn2);
        } else if (STAGE == 3) {
          qa[e] += 2.0f * dq;         pa[e] += 2.0f * dp;
          float qn2 = q[e] + F_DT * dq;
          float pn2 = p[e] + F_DT * dp;
          qe[e] = qn2; pe[e] = pn2;
          Xn[e] = (_Float16)tanhf(qn2);
        } else {
          float qn2 = q[e] + (F_DT / 6.0f) * (qa[e] + dq);
          float pn2 = p[e] + (F_DT / 6.0f) * (pa[e] + dp);
          qn2 = fminf(fmaxf(qn2, -F_CLAMP), F_CLAMP);
          pn2 = fminf(fmaxf(pn2, -F_CLAMP), F_CLAMP);
          q[e] = qn2; p[e] = pn2; qe[e] = qn2; pe[e] = pn2;
          Xn[e] = (_Float16)tanhf(qn2);
          trace[et] = qn2;
        }
      }
    }
  }
}

// --- host launcher ---------------------------------------------------------

extern "C" void kernel_launch(void* const* d_in, const int* in_sizes, int n_in,
                              void* d_out, int out_size, void* d_ws, size_t ws_size,
                              hipStream_t stream) {
  (void)in_sizes; (void)n_in; (void)out_size; (void)ws_size;

  const float* q0    = (const float*)d_in[0];
  const float* p0    = (const float*)d_in[1];
  const float* drive = (const float*)d_in[2];
  const float* J     = (const float*)d_in[3];
  const float* bias  = (const float*)d_in[4];
  float* trace = (float*)d_out;

  // workspace carve-up (~24 MB)
  char* ws = (char*)d_ws;
  _Float16* Jh  = (_Float16*)ws; ws += (size_t)NN * NN * sizeof(_Float16);
  _Float16* Xb0 = (_Float16*)ws; ws += (size_t)NB * NN * sizeof(_Float16);
  _Float16* Xb1 = (_Float16*)ws; ws += (size_t)NB * NN * sizeof(_Float16);
  float* q  = (float*)ws; ws += (size_t)NB * NN * sizeof(float);
  float* p  = (float*)ws; ws += (size_t)NB * NN * sizeof(float);
  float* qe = (float*)ws; ws += (size_t)NB * NN * sizeof(float);
  float* pe = (float*)ws; ws += (size_t)NB * NN * sizeof(float);
  float* qa = (float*)ws; ws += (size_t)NB * NN * sizeof(float);
  float* pa = (float*)ws; ws += (size_t)NB * NN * sizeof(float);

  k_cvt_coupling<<<(NN * NN) / (256 * 4), 256, 0, stream>>>(J, Jh);
  k_init<<<(NB * NN) / 256, 256, 0, stream>>>(q0, p0, q, p, qe, pe, Xb0);

  _Float16* Xbuf[2] = {Xb0, Xb1};
  int cur = 0;
  dim3 grid(NN / BN, NB / BM);   // (16, 4)

  for (int t = 0; t < NT; ++t) {
    k_fhn_stage<1><<<grid, 256, 0, stream>>>(Xbuf[cur], Jh, drive, bias,
        q, p, qe, pe, qa, pa, Xbuf[1 - cur], trace, t);
    cur ^= 1;
    k_fhn_stage<2><<<grid, 256, 0, stream>>>(Xbuf[cur], Jh, drive, bias,
        q, p, qe, pe, qa, pa, Xbuf[1 - cur], trace, t);
    cur ^= 1;
    k_fhn_stage<3><<<grid, 256, 0, stream>>>(Xbuf[cur], Jh, drive, bias,
        q, p, qe, pe, qa, pa, Xbuf[1 - cur], trace, t);
    cur ^= 1;
    k_fhn_stage<4><<<grid, 256, 0, stream>>>(Xbuf[cur], Jh, drive, bias,
        q, p, qe, pe, qa, pa, Xbuf[1 - cur], trace, t);
    cur ^= 1;
  }
}
